// NonOverlappingConv1d_70858370450180
// MI455X (gfx1250) — matrix operand order repaired and checked by reading, not compiled, not run on hardware
//
#include <hip/hip_runtime.h>
#include <stdint.h>

// ---- CDNA5 WMMA types ------------------------------------------------------
typedef __bf16    v16bf __attribute__((ext_vector_type(16)));
typedef float     v8f   __attribute__((ext_vector_type(8)));
typedef uint32_t  u32x4 __attribute__((ext_vector_type(4)));

union FragBF {            // 32 bytes = 8 VGPRs = one 16-bit WMMA operand
    u32x4 q[2];
    v16bf bf;
};

// scheduling-group masks (AMDGPU sched_group_barrier)
#define SG_WMMA    0x008   // MFMA/WMMA
#define SG_DSREAD  0x100   // DS read
#if __has_builtin(__builtin_amdgcn_sched_group_barrier)
#define SCHED_GROUP(mask, size, id) __builtin_amdgcn_sched_group_barrier(mask, size, id)
#else
#define SCHED_GROUP(mask, size, id)
#endif

// ---- problem constants (from reference) ------------------------------------
#define CIN    128
#define D      8192
#define P      4096      // D/2
#define COUT   128
#define BATCH  32

// ---- tiling ----------------------------------------------------------------
#define TILE_M  64       // output channels per WG (4 waves x 16)
#define TILE_N  128      // output positions per WG
#define TILE_K  128      // K chunk (of 256 total), 2 chunks
#define ROWB    272      // 128 bf16 = 256B + 16B pad (16B aligned, bank spread)
#define W_LDS   (TILE_M * ROWB)   // 17408 B
#define X_LDS   (TILE_N * ROWB)   // 34816 B

// round-to-nearest-even f32 -> bf16, packed pair into one dword
__device__ __forceinline__ uint32_t pack_bf16(float lo, float hi) {
    uint32_t a = __float_as_uint(lo);
    uint32_t b = __float_as_uint(hi);
    a = (a + 0x7FFFu + ((a >> 16) & 1u)) >> 16;
    b = (b + 0x7FFFu + ((b >> 16) & 1u)) >> 16;
    return (b << 16) | (a & 0xFFFFu);
}

// A fragment (16x32 bf16): lane<16 -> K[0..7],[16..23]; lane>=16 shifted by +8
__device__ __forceinline__ void loadA(FragBF& f, const uint8_t* sW,
                                      int row, int k0b, int sel) {
    const uint8_t* p = sW + row * ROWB + k0b + sel * 16;
    f.q[0] = *(const u32x4*)(p);
    f.q[1] = *(const u32x4*)(p + 32);
}

// B fragment (32x16 bf16): lane<16 -> K[0..7],[16..23]; lane>=16 shifted by +8
__device__ __forceinline__ void loadB(FragBF& f, const uint8_t* sX,
                                      int nrow, int k0b, int sel) {
    const uint8_t* p = sX + nrow * ROWB + k0b + sel * 16;
    f.q[0] = *(const u32x4*)(p);
    f.q[1] = *(const u32x4*)(p + 32);
}

__global__ __launch_bounds__(128)
void NonOverlappingConv1d_wmma_kernel(const float* __restrict__ x,
                                      const float* __restrict__ w,
                                      float* __restrict__ out)
{
    __shared__ __align__(16) uint8_t smem[W_LDS + X_LDS];   // 52224 B
    uint8_t* sW = smem;
    uint8_t* sX = smem + W_LDS;

    const int tid  = threadIdx.x;
    const int lane = tid & 31;
    const int wv   = tid >> 5;          // wave id 0..3  -> M rows [wv*16, wv*16+16)
    const int l16  = lane & 15;
    const int sel  = lane >> 4;         // 0: lanes 0-15, 1: lanes 16-31

    const int p0 = blockIdx.x * TILE_N; // output-position tile
    const int m0 = blockIdx.y * TILE_M; // output-channel tile
    const int b  = blockIdx.z;          // batch

    const float* xb = x + (size_t)b * CIN * D;
    const int arow = wv * 16 + l16;     // local A row for this lane (constant)

    v8f acc[8];
    #pragma unroll
    for (int i = 0; i < 8; ++i)
        #pragma unroll
        for (int j = 0; j < 8; ++j)
            acc[i][j] = 0.0f;

    for (int chunk = 0; chunk < 2; ++chunk) {
        __syncthreads();   // LDS reuse guard

        // ---- W tile: rows m0..m0+63, kk in [chunk*128, chunk*128+128) ------
        // w is already [o][kk=2c+k] contiguous: o*256 + kk.
        #pragma unroll 4
        for (int i = 0; i < 16; ++i) {
            int flat = i * 128 + tid;            // 0..2047
            int o    = flat >> 5;                // local row 0..63
            int q    = flat & 31;                // kk group-of-4 within chunk
            const float4 v =
                *(const float4*)(w + (size_t)(m0 + o) * 256 + chunk * TILE_K + q * 4);
            uint32_t d0 = pack_bf16(v.x, v.y);
            uint32_t d1 = pack_bf16(v.z, v.w);
            uint32_t* dst = (uint32_t*)(sW + o * ROWB + q * 8);
            dst[0] = d0;
            dst[1] = d1;
        }

        // ---- X tile (transposed to N-major/K-contig): ----------------------
        // LDS row p holds K = 2*cl + k for cl in [0,64).  Global source:
        // x[b, c, 2p + k], 4 consecutive f32 cover (p, p+1) x (k=0,1).
        #pragma unroll 4
        for (int i = 0; i < 32; ++i) {
            int flat = i * 128 + tid;            // 0..4095
            int cl   = flat >> 6;                // local c 0..63
            int pg   = flat & 63;                // pair-of-p group
            int c    = chunk * 64 + cl;
            const float4 v =
                *(const float4*)(xb + (size_t)c * D + 2 * p0 + pg * 4);
            uint32_t d0 = pack_bf16(v.x, v.y);   // row 2*pg,   K=2cl..2cl+1
            uint32_t d1 = pack_bf16(v.z, v.w);   // row 2*pg+1, K=2cl..2cl+1
            int prow = pg * 2;
            *(uint32_t*)(sX + prow * ROWB + cl * 4)       = d0;
            *(uint32_t*)(sX + (prow + 1) * ROWB + cl * 4) = d1;
        }

        __syncthreads();

        // ---- compute: 4 K-steps of 32, 8 N-fragments each ------------------
        // Pair-wise software pipeline: 4 live B buffers (two ping-pong pairs)
        // so the loads for fragments nf+2, nf+3 are in flight while the WMMAs
        // for nf, nf+1 execute.
        FragBF afr[2], bfr[4];
        loadA(afr[0], sW, arow, 0, sel);
        loadB(bfr[0], sX, l16, 0, sel);
        loadB(bfr[1], sX, 16 + l16, 0, sel);

        #pragma unroll
        for (int ks = 0; ks < 4; ++ks) {
            const int k0b = ks * 64;             // K base in bytes (32 K * 2B)
            #pragma unroll
            for (int nfp = 0; nfp < 4; ++nfp) {  // pairs of N-fragments
                const int base = nfp * 2;
                const int curp = (nfp & 1) * 2;  // bfr[curp], bfr[curp+1]
                const int nxtp = curp ^ 2;
                if (nfp < 3) {
                    // prefetch next fragment pair of this K-step
                    loadB(bfr[nxtp],     sX, (base + 2) * 16 + l16, k0b, sel);
                    loadB(bfr[nxtp + 1], sX, (base + 3) * 16 + l16, k0b, sel);
                } else if (ks < 3) {
                    // prefetch A + first fragment pair of next K-step
                    const int nk0b = (ks + 1) * 64;
                    loadA(afr[(ks + 1) & 1], sW, arow, nk0b, sel);
                    loadB(bfr[nxtp],     sX, l16,      nk0b, sel);
                    loadB(bfr[nxtp + 1], sX, 16 + l16, nk0b, sel);
                }
                acc[base] = __builtin_amdgcn_wmma_f32_16x16x32_bf16(
                    false, afr[ks & 1].bf, false, bfr[curp].bf,
                    (short)0, acc[base], false, false);
                acc[base + 1] = __builtin_amdgcn_wmma_f32_16x16x32_bf16(
                    false, afr[ks & 1].bf, false, bfr[curp + 1].bf,
                    (short)0, acc[base + 1], false, false);
            }
        }

        // ---- forced interleave for this scheduling region ------------------
        // 8 DS-reads up front (A + B pair 0 + half of pair 1), then a
        // steady state of 1 WMMA : 2 DS-reads.  Keeps load issue ~2 fragment
        // pairs ahead of consumption -> waits become dscnt<=4 instead of 0.
        SCHED_GROUP(SG_DSREAD, 8, 0);
        #pragma unroll
        for (int i = 0; i < 32; ++i) {
            SCHED_GROUP(SG_WMMA,   1, 0);
            SCHED_GROUP(SG_DSREAD, 2, 0);
        }
    }

    // ---- epilogue: scale by 1/sqrt(128), store C layout ---------------------
    const float scale = 0.08838834764831845f;
    float* ob = out + (size_t)b * COUT * P;
    #pragma unroll
    for (int nf = 0; nf < 8; ++nf) {
        const int pcol = p0 + nf * 16 + l16;
        #pragma unroll
        for (int v = 0; v < 8; ++v) {
            const int orow = m0 + wv * 16 + sel * 8 + v;   // VGPR v -> M = 8*sel + v
            ob[(size_t)orow * P + pcol] = acc[nf][v] * scale;
        }
    }
}

extern "C" void kernel_launch(void* const* d_in, const int* in_sizes, int n_in,
                              void* d_out, int out_size, void* d_ws, size_t ws_size,
                              hipStream_t stream) {
    (void)in_sizes; (void)n_in; (void)out_size; (void)d_ws; (void)ws_size;
    const float* x = (const float*)d_in[0];   // (32, 128, 8192) f32
    const float* w = (const float*)d_in[1];   // (128, 128, 1, 2) f32
    float* out     = (float*)d_out;           // (32, 128, 4096) f32

    dim3 grid(P / TILE_N, COUT / TILE_M, BATCH);   // (32, 2, 32)
    dim3 block(128);
    NonOverlappingConv1d_wmma_kernel<<<grid, block, 0, stream>>>(x, w, out);
}
